// GatedDeltaNetAttention_53352083751003
// MI455X (gfx1250) — matrix operand order repaired and checked by reading, not compiled, not run on hardware
//
#include <hip/hip_runtime.h>
#include <hip/hip_bf16.h>

typedef _Float16 v8h  __attribute__((ext_vector_type(8)));
typedef _Float16 v16h __attribute__((ext_vector_type(16)));
typedef float    v8f  __attribute__((ext_vector_type(8)));
typedef int      v4i  __attribute__((vector_size(16)));

#define D_MODEL 1024
#define NHEAD   16
#define DHEAD   64
#define SEQ     2048
#define NBATCH  2
#define MROWS   (NBATCH * SEQ)

#ifndef __has_builtin
#define __has_builtin(x) 0
#endif

#if defined(__AMDGCN__) && __has_builtin(__builtin_amdgcn_global_load_async_to_lds_b128)
#define USE_ASYNC_LDS 1
#else
#define USE_ASYNC_LDS 0
#endif

typedef __attribute__((address_space(1))) v4i as1_v4i;
typedef __attribute__((address_space(3))) v4i as3_v4i;

// 16-byte global -> LDS copy. Async (ASYNCcnt) on gfx1250 when available.
__device__ __forceinline__ void copy16(const _Float16* g, _Float16* l) {
#if USE_ASYNC_LDS
  __builtin_amdgcn_global_load_async_to_lds_b128(
      (as1_v4i*)(void*)g, (as3_v4i*)(void*)l, 0, 0);
#else
  *(v8h*)l = *(const v8h*)g;
#endif
}

__device__ __forceinline__ void wait_async() {
#if USE_ASYNC_LDS
#if __has_builtin(__builtin_amdgcn_s_wait_asynccnt)
  __builtin_amdgcn_s_wait_asynccnt(0);
#else
  asm volatile("s_wait_asynccnt 0x0" ::: "memory");
#endif
#endif
}

// ---------------------------------------------------------------------------
// WMMA fragment loaders (CDNA5 ISA 7.12.2 layouts, wave32)
// A 16x32 f16: lane<16 -> row=lane, K[0..7]+K[16..23]; lane>=16 -> K[8..15]+K[24..31]
// ---------------------------------------------------------------------------
__device__ __forceinline__ v16h frag_a(const _Float16* p0, int stride) {
  int lane = threadIdx.x & 31;
  const _Float16* p = p0 + (lane & 15) * stride + ((lane >> 4) << 3);
  v8h lo = *(const v8h*)p;
  v8h hi = *(const v8h*)(p + 16);
  v16h a;
#pragma unroll
  for (int i = 0; i < 8; ++i) { a[i] = lo[i]; a[i + 8] = hi[i]; }
  return a;
}

// B 32x16 f16 from transposed LDS tile Bt[n][k] (k contiguous)
__device__ __forceinline__ v16h frag_b(const _Float16* p0, int stride) {
  int lane = threadIdx.x & 31;
  const _Float16* p = p0 + (lane & 15) * stride + ((lane >> 4) << 3);
  v8h lo = *(const v8h*)p;
  v8h hi = *(const v8h*)(p + 16);
  v16h b;
#pragma unroll
  for (int i = 0; i < 8; ++i) { b[i] = lo[i]; b[i + 8] = hi[i]; }
  return b;
}

// ---------------------------------------------------------------------------
// WMMA GEMM: C[M,N] = A[M,K] (f16 row-major) x Bt[N,K] (f16, B transposed).
// 128x128 block, 8 waves (2x4), wave tile 64x32, BK=32, double-buffered LDS
// with async global->LDS staging pipelined against the WMMA mainloop.
// MODE: 0 = f32 out, 1 = f16 out, 2 = f16 out in [b,h,e,l] transposed layout.
// ---------------------------------------------------------------------------
template <int MODE>
__global__ __launch_bounds__(256) void gemm_wmma(
    const _Float16* __restrict__ A, const _Float16* __restrict__ Bt,
    void* __restrict__ C, int M, int N, int K) {
  __shared__ __align__(16) _Float16 sA[2][128][40];
  __shared__ __align__(16) _Float16 sB[2][128][40];

  const int tid  = threadIdx.x;
  const int wid  = tid >> 5, lane = tid & 31;
  const int wm   = (wid >> 2) * 64;
  const int wn   = (wid & 3) * 32;
  const int bm   = blockIdx.y * 128;
  const int bn   = blockIdx.x * 128;
  const int cn   = lane & 15;
  const int rsel = (lane >> 4) * 8;

  auto issue_tile = [&](int buf, int k0) {
    for (int c = tid; c < 512; c += 256) {
      int r = c >> 2, kc = (c & 3) * 8;
      copy16(&A[(size_t)(bm + r) * K + k0 + kc], &sA[buf][r][kc]);
      copy16(&Bt[(size_t)(bn + r) * K + k0 + kc], &sB[buf][r][kc]);
    }
  };

  v8f acc[4][2] = {};

  const int KT = K >> 5;
  issue_tile(0, 0);
  for (int kt = 0; kt < KT; ++kt) {
    const int cur = kt & 1;
    wait_async();        // my chunks of tile kt have landed in LDS
    __syncthreads();     // everyone's chunks landed; prev compute done
    if (kt + 1 < KT) issue_tile(cur ^ 1, (kt + 1) << 5);  // prefetch next

    v16h bf[2];
#pragma unroll
    for (int n = 0; n < 2; ++n) bf[n] = frag_b(&sB[cur][wn + n * 16][0], 40);
#pragma unroll
    for (int m = 0; m < 4; ++m) {
      v16h af = frag_a(&sA[cur][wm + m * 16][0], 40);
#pragma unroll
      for (int n = 0; n < 2; ++n)
        acc[m][n] = __builtin_amdgcn_wmma_f32_16x16x32_f16(
            false, af, false, bf[n], (short)0, acc[m][n], false, false);
    }
  }

#pragma unroll
  for (int m = 0; m < 4; ++m)
#pragma unroll
    for (int n = 0; n < 2; ++n)
#pragma unroll
      for (int j = 0; j < 8; ++j) {
        int row = bm + wm + m * 16 + j + rsel;
        int col = bn + wn + n * 16 + cn;
        float val = acc[m][n][j];
        if (MODE == 0) {
          ((float*)C)[(size_t)row * N + col] = val;
        } else if (MODE == 1) {
          ((_Float16*)C)[(size_t)row * N + col] = (_Float16)val;
        } else {  // MODE==2: row = b*SEQ+l, col = h*64+e -> [b,h,e,l]
          int bi = row >> 11, l = row & (SEQ - 1);
          int h = col >> 6, e = col & (DHEAD - 1);
          ((_Float16*)C)[(size_t)((bi * NHEAD + h) * DHEAD + e) * SEQ + l] =
              (_Float16)val;
        }
      }
}

// ---------------------------------------------------------------------------
// Causal linear attention per (b, h, 128-row l-block), flash-style.
// out[l,e] = sum_{t<=l} beta[t] * <q[l],k[t]> * v[t,e]
// q,k: [b,l,h,d] f16 (normalized). vT: [b,h,e,t] f16. All tiles async-staged.
// ---------------------------------------------------------------------------
__global__ __launch_bounds__(256) void attn_kernel(
    const _Float16* __restrict__ q, const _Float16* __restrict__ k,
    const _Float16* __restrict__ vT, const float* __restrict__ beta,
    _Float16* __restrict__ out) {
  __shared__ __align__(16) _Float16 qs[128][72];
  __shared__ __align__(16) _Float16 ks[64][72];
  __shared__ __align__(16) _Float16 vts[64][72];   // vts[e][t]
  __shared__ __align__(16) _Float16 ss[8][16][72]; // per-wave scores scratch
  __shared__ float sbeta[64];

  const int tid  = threadIdx.x;
  const int wid  = tid >> 5, lane = tid & 31;
  const int lb   = blockIdx.x;
  const int h    = blockIdx.y;
  const int b    = blockIdx.z;
  const int l0   = lb * 128;
  const int cn   = lane & 15;
  const int rsel = (lane >> 4) * 8;

  // stage q block once (async)
  for (int c = tid; c < 1024; c += 256) {
    int r = c >> 3, d0 = (c & 7) * 8;
    copy16(&q[(size_t)(b * SEQ + l0 + r) * D_MODEL + h * DHEAD + d0],
           &qs[r][d0]);
  }

  v8f o[4] = {};

  const int tbmax = 2 * lb + 1;  // causal: only t-blocks with t0 <= l0+127
  for (int tb = 0; tb <= tbmax; ++tb) {
    const int t0 = tb * 64;
    __syncthreads();  // previous compute's reads of ks/vts are complete
    for (int c = tid; c < 512; c += 256) {
      int r = c >> 3, d0 = (c & 7) * 8;
      copy16(&k[(size_t)(b * SEQ + t0 + r) * D_MODEL + h * DHEAD + d0],
             &ks[r][d0]);
      copy16(&vT[(size_t)((b * NHEAD + h) * DHEAD + r) * SEQ + t0 + d0],
             &vts[r][d0]);
    }
    if (tid < 64)
      sbeta[tid] = beta[(size_t)(b * SEQ + t0 + tid) * NHEAD + h];
    wait_async();     // covers q on first iteration too
    __syncthreads();

    // GEMM1: scores = q . k^T  (16 l-rows per wave x 64 t-cols, K = 64)
    v8f s[4] = {};
#pragma unroll
    for (int kd = 0; kd < 2; ++kd) {
      v16h a = frag_a(&qs[wid * 16][kd * 32], 72);
#pragma unroll
      for (int n = 0; n < 4; ++n) {
        v16h bfr = frag_b(&ks[n * 16][kd * 32], 72);
        s[n] = __builtin_amdgcn_wmma_f32_16x16x32_f16(
            false, a, false, bfr, (short)0, s[n], false, false);
      }
    }

    // beta scale + causal mask; C-layout -> A-layout via per-wave LDS
#pragma unroll
    for (int n = 0; n < 4; ++n) {
      int   t  = t0 + n * 16 + cn;  // column is per-lane constant
      float bt = sbeta[n * 16 + cn];
#pragma unroll
      for (int j = 0; j < 8; ++j) {
        int   l   = l0 + wid * 16 + j + rsel;
        float val = (t <= l) ? s[n][j] * bt : 0.0f;
        ss[wid][j + rsel][n * 16 + cn] = (_Float16)val;
      }
    }

    // GEMM2: o += scores @ v (K = 64). LDS in-order per wave: no barrier.
#pragma unroll
    for (int kt = 0; kt < 2; ++kt) {
      v16h a = frag_a(&ss[wid][0][kt * 32], 72);
#pragma unroll
      for (int n = 0; n < 4; ++n) {
        v16h bfr = frag_b(&vts[n * 16][kt * 32], 72);
        o[n] = __builtin_amdgcn_wmma_f32_16x16x32_f16(
            false, a, false, bfr, (short)0, o[n], false, false);
      }
    }
  }

#pragma unroll
  for (int n = 0; n < 4; ++n)
#pragma unroll
    for (int j = 0; j < 8; ++j) {
      int l = l0 + wid * 16 + j + rsel;
      int e = n * 16 + cn;
      out[(size_t)(b * SEQ + l) * D_MODEL + h * DHEAD + e] = (_Float16)o[n][j];
    }
}

// ---------------------------------------------------------------------------
// Prologue / epilogue helpers
// ---------------------------------------------------------------------------
__global__ void f32_to_f16(const float* __restrict__ in,
                           _Float16* __restrict__ out, int n) {
  int i = blockIdx.x * blockDim.x + threadIdx.x;
  if (i < n) out[i] = (_Float16)in[i];
}

__global__ void transpose_f32_to_f16(const float* __restrict__ W,
                                     _Float16* __restrict__ Wt, int K, int N) {
  __shared__ float tile[32][33];
  int n0 = blockIdx.x * 32, k0 = blockIdx.y * 32;
  int tx = threadIdx.x & 31, ty = threadIdx.x >> 5;
  for (int i = ty; i < 32; i += 8)
    tile[i][tx] = W[(size_t)(k0 + i) * N + n0 + tx];
  __syncthreads();
  for (int i = ty; i < 32; i += 8)
    Wt[(size_t)(n0 + i) * K + k0 + tx] = (_Float16)tile[tx][i];
}

// one wave32 per 64-element head row
__global__ void l2norm_rows64(_Float16* __restrict__ t, int nrows) {
  int row  = (blockIdx.x * blockDim.x + threadIdx.x) >> 5;
  int lane = threadIdx.x & 31;
  if (row >= nrows) return;
  _Float16* p = t + (size_t)row * 64;
  float a  = (float)p[lane];
  float b2 = (float)p[lane + 32];
  float ss = a * a + b2 * b2;
#pragma unroll
  for (int off = 16; off > 0; off >>= 1) ss += __shfl_xor(ss, off, 32);
  float inv = 1.0f / fmaxf(sqrtf(ss), 1e-12f);
  p[lane]      = (_Float16)(a * inv);
  p[lane + 32] = (_Float16)(b2 * inv);
}

__global__ void beta_kernel(const float* __restrict__ x,
                            const float* __restrict__ Wb,
                            const float* __restrict__ bb,
                            float* __restrict__ beta, int rows) {
  int idx = blockIdx.x * blockDim.x + threadIdx.x;
  if (idx >= rows * NHEAD) return;
  int r = idx >> 4, h = idx & 15;
  const float* xr = x + (size_t)r * D_MODEL;
  float s = bb[h];
  for (int kk = 0; kk < D_MODEL; ++kk) s += xr[kk] * Wb[kk * NHEAD + h];
  beta[idx] = 1.0f / (1.0f + __expf(-s));
}

// ---------------------------------------------------------------------------
// Launch
// ---------------------------------------------------------------------------
extern "C" void kernel_launch(void* const* d_in, const int* in_sizes, int n_in,
                              void* d_out, int out_size, void* d_ws,
                              size_t ws_size, hipStream_t stream) {
  const float* x  = (const float*)d_in[0];
  const float* Wq = (const float*)d_in[1];
  const float* Wk = (const float*)d_in[2];
  const float* Wv = (const float*)d_in[3];
  const float* Wo = (const float*)d_in[4];
  const float* Wb = (const float*)d_in[5];
  const float* bb = (const float*)d_in[6];

  char*  ws  = (char*)d_ws;
  size_t off = 0;
  auto alloc = [&](size_t bytes) -> char* {
    char* p = ws + off;
    off += (bytes + 255) & ~(size_t)255;
    return p;
  };
  _Float16* x16  = (_Float16*)alloc((size_t)MROWS * D_MODEL * 2);
  _Float16* Wqt  = (_Float16*)alloc((size_t)D_MODEL * D_MODEL * 2);
  _Float16* Wkt  = (_Float16*)alloc((size_t)D_MODEL * D_MODEL * 2);
  _Float16* Wvt  = (_Float16*)alloc((size_t)D_MODEL * D_MODEL * 2);
  _Float16* Wot  = (_Float16*)alloc((size_t)D_MODEL * D_MODEL * 2);
  _Float16* q16  = (_Float16*)alloc((size_t)MROWS * D_MODEL * 2);
  _Float16* k16  = (_Float16*)alloc((size_t)MROWS * D_MODEL * 2);
  _Float16* vT16 = (_Float16*)alloc((size_t)MROWS * D_MODEL * 2);  // [b,h,e,t]
  _Float16* a16  = (_Float16*)alloc((size_t)MROWS * D_MODEL * 2);
  float*    beta = (float*)alloc((size_t)MROWS * NHEAD * 4);

  // prologue: convert x, convert+transpose weights
  int nx = MROWS * D_MODEL;
  f32_to_f16<<<(nx + 255) / 256, 256, 0, stream>>>(x, x16, nx);
  dim3 tg(D_MODEL / 32, D_MODEL / 32);
  transpose_f32_to_f16<<<tg, 256, 0, stream>>>(Wq, Wqt, D_MODEL, D_MODEL);
  transpose_f32_to_f16<<<tg, 256, 0, stream>>>(Wk, Wkt, D_MODEL, D_MODEL);
  transpose_f32_to_f16<<<tg, 256, 0, stream>>>(Wv, Wvt, D_MODEL, D_MODEL);
  transpose_f32_to_f16<<<tg, 256, 0, stream>>>(Wo, Wot, D_MODEL, D_MODEL);

  // projections (WMMA, async-pipelined); V written directly transposed
  dim3 gg(D_MODEL / 128, MROWS / 128);
  gemm_wmma<1><<<gg, 256, 0, stream>>>(x16, Wqt, q16, MROWS, D_MODEL, D_MODEL);
  gemm_wmma<1><<<gg, 256, 0, stream>>>(x16, Wkt, k16, MROWS, D_MODEL, D_MODEL);
  gemm_wmma<2><<<gg, 256, 0, stream>>>(x16, Wvt, vT16, MROWS, D_MODEL, D_MODEL);

  // normalize q, k per head row; compute beta gates
  int nrows = MROWS * NHEAD;
  l2norm_rows64<<<nrows / 8, 256, 0, stream>>>(q16, nrows);
  l2norm_rows64<<<nrows / 8, 256, 0, stream>>>(k16, nrows);
  beta_kernel<<<(nrows + 255) / 256, 256, 0, stream>>>(x, Wb, bb, beta, MROWS);

  // causal gated attention (WMMA + async staging)
  attn_kernel<<<dim3(SEQ / 128, NHEAD, NBATCH), 256, 0, stream>>>(
      q16, k16, vT16, beta, a16);

  // output projection -> f32 d_out (WMMA)
  gemm_wmma<0><<<gg, 256, 0, stream>>>(a16, Wot, (float*)d_out, MROWS,
                                       D_MODEL, D_MODEL);
}
